// InformationPrototype_23493471109706
// MI455X (gfx1250) — matrix operand order repaired and checked by reading, not compiled, not run on hardware
//
#include <hip/hip_runtime.h>
#include <hip/hip_bf16.h>

typedef __attribute__((ext_vector_type(16))) _Float16 v16h;
typedef __attribute__((ext_vector_type(8)))  float    v8f;
typedef __attribute__((__vector_size__(16))) int      v4i;   // 4 x i32 payload

// Problem sizes (fixed by the reference)
#define B_ROIS 2048
#define DFEAT  2048
#define NCLS   21
#define HWSZ   49
#define EPS_F  1e-8f

// Flat output layout (tuple concatenated in return order, all stored as f32):
//   new_prototypes [21,2048]  step [21]  x [2048,2048,7,7]
//   class_logits [2048,21]    max_cls [2048]   x_mapped [2048,2048]
#define OFF_PROTO   ((size_t)0)
#define OFF_STEP    ((size_t)NCLS * DFEAT)                    // 43008
#define OFF_X       (OFF_STEP + NCLS)                         // 43029
#define OFF_LOGITS  (OFF_X + (size_t)B_ROIS * DFEAT * HWSZ)   // 205563925
#define OFF_MAXCLS  (OFF_LOGITS + (size_t)B_ROIS * NCLS)      // 205606933
#define OFF_XMAP    (OFF_MAXCLS + (size_t)B_ROIS)             // 205608981

// Workspace layout
#define WS_MAXCLS_OFF 0                      // int[2048]
#define WS_COUNTS_OFF (B_ROIS * 4)           // int[32]
#define WS_LOCAL_OFF  (B_ROIS * 4 + 128)     // float[21*2048]

// gfx1250 async LDS<->global data movers (probe via __has_builtin; fall back
// to plain VALU streaming if this toolchain doesn't declare them).
#if defined(__AMDGCN__) && \
    __has_builtin(__builtin_amdgcn_global_load_async_to_lds_b128) && \
    __has_builtin(__builtin_amdgcn_global_store_async_from_lds_b32)
#define HAVE_ASYNC_LDS 1
#else
#define HAVE_ASYNC_LDS 0
#endif

#define AS1 __attribute__((address_space(1)))
#define AS3 __attribute__((address_space(3)))

__device__ __forceinline__ void wait_asynccnt0() {
#if defined(__AMDGCN__)
#if __has_builtin(__builtin_amdgcn_s_wait_asynccnt)
    __builtin_amdgcn_s_wait_asynccnt(0);
#else
    asm volatile("s_wait_asynccnt 0x0" ::: "memory");
#endif
#endif
}

// ---------------------------------------------------------------------------
// Kernel 0: zero the per-class counts (ws is not re-poisoned between replays,
// so we must re-init every launch for determinism).
__global__ void k_zero_counts(int* counts) {
    if (threadIdx.x < 32) counts[threadIdx.x] = 0;
}

// ---------------------------------------------------------------------------
// Kernel 1: argmax over logits (softmax is monotone -> argmax(logits)),
// copy logits through, emit max_cls (f32 flat output + int workspace),
// histogram counts, and pass `step` through.
__global__ void k_argmax(const float* __restrict__ logits,
                         const int* __restrict__ step,
                         float* __restrict__ out,
                         int* __restrict__ ws_maxcls,
                         int* __restrict__ ws_counts) {
    int b = blockIdx.x * blockDim.x + threadIdx.x;
    if (b < B_ROIS) {
        const float* row = logits + (size_t)b * NCLS;
        float* orow = out + OFF_LOGITS + (size_t)b * NCLS;
        float best = row[0];
        int bi = 0;
        orow[0] = best;
        #pragma unroll
        for (int c = 1; c < NCLS; ++c) {
            float v = row[c];
            orow[c] = v;
            if (v > best) { best = v; bi = c; }
        }
        ws_maxcls[b] = bi;
        atomicAdd(&ws_counts[bi], 1);
        out[OFF_MAXCLS + b] = (float)bi;
    }
    if (blockIdx.x == 0 && threadIdx.x < NCLS) {
        out[OFF_STEP + threadIdx.x] = (float)step[threadIdx.x];
    }
}

// ---------------------------------------------------------------------------
// Kernel 2: fused AvgPool2d(7,7) + identity passthrough of x.
// Reads the 822MB tensor exactly once (~1.65 GB total traffic -> ~71us at
// 23.3 TB/s). Async-LDS variant: stage a 256-row tile (50176 B) in LDS with
// coalesced global_load_async_to_lds_b128 (tile base 16B-aligned since
// row0*196 is a multiple of 50176), reduce rows from LDS (49 elems/row;
// gcd(49,64)=1 -> the 32 lanes of a wave hit 32 distinct banks), then stream
// the passthrough copy out with global_store_async_from_lds_b32 (x-output
// region is only 4B-aligned at flat offset 43029).
#define TILE_ROWS   256
#define TILE_FLOATS (TILE_ROWS * HWSZ)   // 12544 floats = 50176 B LDS
#define TILE_VEC4   (TILE_FLOATS / 4)    // 3136 16-byte chunks

__global__ void k_pool_copy(const float* __restrict__ x,
                            float* __restrict__ out) {
#if HAVE_ASYNC_LDS
    __shared__ float tile[TILE_FLOATS];
    const int tid = threadIdx.x;                       // 256 threads = 8 waves
    const size_t row0 = (size_t)blockIdx.x * TILE_ROWS;

    AS1 const char* gsrc = (AS1 const char*)(const void*)(x + row0 * HWSZ);
    AS3 char*       lbuf = (AS3 char*)(void*)&tile[0];

    // Phase 1: async-stage the tile into LDS, 16 B per lane per op, coalesced.
    for (int j = tid; j < TILE_VEC4; j += 256) {
        __builtin_amdgcn_global_load_async_to_lds_b128(
            (AS1 v4i*)(gsrc + (size_t)j * 16), (AS3 v4i*)(lbuf + j * 16),
            /*offset=*/0, /*cpol=*/0);
    }
    wait_asynccnt0();
    __syncthreads();

    // Phase 2: per-row mean from LDS (bank-conflict-free across a wave).
    float s = 0.0f;
    #pragma unroll
    for (int i = 0; i < HWSZ; ++i) s += tile[tid * HWSZ + i];
    out[OFF_XMAP + row0 + tid] = s * (1.0f / (float)HWSZ);

    // Phase 3: async-stream the identity copy LDS -> global, coalesced b32.
    AS1 char* gdst = (AS1 char*)(void*)(out + OFF_X + row0 * HWSZ);
    for (int e = tid; e < TILE_FLOATS; e += 256) {
        __builtin_amdgcn_global_store_async_from_lds_b32(
            (AS1 int*)(gdst + (size_t)e * 4), (AS3 int*)(lbuf + e * 4),
            /*offset=*/0, /*cpol=*/0);
    }
    wait_asynccnt0();
#else
    // Fallback: per-thread 49-float streaming with gfx1250 prefetch.
    const size_t nrows = (size_t)B_ROIS * DFEAT;
    size_t stride = (size_t)gridDim.x * blockDim.x;
    float* out_x    = out + OFF_X;
    float* out_xmap = out + OFF_XMAP;
    for (size_t row = (size_t)blockIdx.x * blockDim.x + threadIdx.x;
         row < nrows; row += stride) {
        size_t base = row * HWSZ;
        if (row + stride < nrows)
            __builtin_prefetch(&x[(row + stride) * HWSZ], 0, 0);
        float s = 0.0f;
        #pragma unroll
        for (int i = 0; i < HWSZ; ++i) {
            float v = x[base + i];
            out_x[base + i] = v;
            s += v;
        }
        out_xmap[row] = s * (1.0f / (float)HWSZ);
    }
#endif
}

// ---------------------------------------------------------------------------
// Kernel 3: segment-sum as a one-hot GEMM on the WMMA pipe.
//   local[C, D] = onehot[C, B] @ x_mapped[B, D],  C padded 21 -> 32.
// One wave per 16x16 output tile; K loop over B in steps of 32 using
// v_wmma_f32_16x16x32_f16. One-hot entries are exact in f16.
// VGPR packing used here (wave32, contiguous k for A and B):
//   A 16x32 f16 : M = lane&15; elem i -> K = i + (lane>=16?16:0)
//   B 32x16 f16 : N = lane&15; elem i -> K = i + (lane>=16?16:0)
//   C 16x16 f32 : elem g -> M = g + (lane>=16?8:0), N = lane&15
__global__ void k_segsum_wmma(const float* __restrict__ xmap,
                              const int* __restrict__ ws_maxcls,
                              float* __restrict__ ws_local) {
    __shared__ int smc[B_ROIS];
    int lane = threadIdx.x;              // block = 1 wave of 32
    for (int i = lane; i < B_ROIS; i += 32) smc[i] = ws_maxcls[i];
    __syncthreads();

    const int laneLo = lane & 15;
    const bool hi    = lane >= 16;
    const int mbase  = blockIdx.y * 16;             // class tile (0 or 16)
    const int col    = blockIdx.x * 16 + laneLo;    // feature column
    const int myM    = mbase + laneLo;              // one-hot row this lane fills

    v8f acc = {};
    for (int k0 = 0; k0 < B_ROIS; k0 += 32) {
        v16h a, b;
        #pragma unroll
        for (int i = 0; i < 16; ++i) {
            int ka = k0 + i + (hi ? 16 : 0);
            a[i] = (_Float16)((smc[ka] == myM) ? 1.0f : 0.0f);
            int kb = k0 + i + (hi ? 16 : 0);
            b[i] = (_Float16)xmap[(size_t)kb * DFEAT + col];
        }
        acc = __builtin_amdgcn_wmma_f32_16x16x32_f16(
            /*neg_a=*/false, a, /*neg_b=*/false, b,
            /*c_mod=*/(short)0, acc, /*reuse_a=*/false, /*reuse_b=*/false);
    }

    #pragma unroll
    for (int g = 0; g < 8; ++g) {
        int m = mbase + g + (hi ? 8 : 0);
        if (m < NCLS) ws_local[(size_t)m * DFEAT + col] = acc[g];
    }
}

// ---------------------------------------------------------------------------
// Kernel 4: per-class cosine similarity + momentum blend.
// One block per class; 3-way block reduction over D=2048.
__global__ void k_update(const float* __restrict__ proto,
                         const float* __restrict__ ws_local,
                         const int* __restrict__ ws_counts,
                         float* __restrict__ out) {
    __shared__ float s_dot[256], s_na[256], s_nb[256];
    __shared__ float s_mom;
    int c = blockIdx.x;
    int t = threadIdx.x;
    int cnt = ws_counts[c];
    bool exist = cnt > 0;
    float invc = 1.0f / (float)(cnt > 0 ? cnt : 1);

    const float* pr = proto    + (size_t)c * DFEAT;
    const float* lo = ws_local + (size_t)c * DFEAT;

    float dot = 0.f, na = 0.f, nb = 0.f;
    for (int d = t; d < DFEAT; d += 256) {
        float p  = pr[d];
        float lm = lo[d] * invc;
        dot += p * lm;
        na  += p * p;
        nb  += lm * lm;
    }
    s_dot[t] = dot; s_na[t] = na; s_nb[t] = nb;
    __syncthreads();
    for (int off = 128; off > 0; off >>= 1) {
        if (t < off) {
            s_dot[t] += s_dot[t + off];
            s_na[t]  += s_na[t + off];
            s_nb[t]  += s_nb[t + off];
        }
        __syncthreads();
    }
    if (t == 0) {
        float da = sqrtf(s_na[0]); if (da < EPS_F) da = EPS_F;
        float db = sqrtf(s_nb[0]); if (db < EPS_F) db = EPS_F;
        s_mom = s_dot[0] / (da * db);
    }
    __syncthreads();
    float mom = s_mom;
    float* outp = out + OFF_PROTO + (size_t)c * DFEAT;
    for (int d = t; d < DFEAT; d += 256) {
        float p  = pr[d];
        float lm = lo[d] * invc;
        outp[d] = exist ? (p * mom + lm * (1.0f - mom)) : p;
    }
}

// ---------------------------------------------------------------------------
extern "C" void kernel_launch(void* const* d_in, const int* in_sizes, int n_in,
                              void* d_out, int out_size, void* d_ws, size_t ws_size,
                              hipStream_t stream) {
    const float* x      = (const float*)d_in[0];
    const float* logits = (const float*)d_in[1];
    const float* proto  = (const float*)d_in[2];
    const int*   step   = (const int*)d_in[3];
    float* out = (float*)d_out;

    char* ws = (char*)d_ws;
    int*   ws_maxcls = (int*)(ws + WS_MAXCLS_OFF);
    int*   ws_counts = (int*)(ws + WS_COUNTS_OFF);
    float* ws_local  = (float*)(ws + WS_LOCAL_OFF);

    // 0) deterministic re-init of counts
    k_zero_counts<<<1, 32, 0, stream>>>(ws_counts);

    // 1) argmax / logits passthrough / step passthrough
    k_argmax<<<(B_ROIS + 255) / 256, 256, 0, stream>>>(
        logits, step, out, ws_maxcls, ws_counts);

    // 2) fused pool + x passthrough (bandwidth-dominant stage)
    //    4M rows / 256-row tiles = 16384 workgroups (exact, no tail)
    k_pool_copy<<<(B_ROIS * DFEAT) / TILE_ROWS, 256, 0, stream>>>(x, out);

    // 3) one-hot GEMM segment-sum on WMMA (2 class tiles x 128 feature tiles)
    dim3 g3(DFEAT / 16, 2);
    k_segsum_wmma<<<g3, 32, 0, stream>>>(out + OFF_XMAP, ws_maxcls, ws_local);

    // 4) cosine-momentum prototype update
    k_update<<<NCLS, 256, 0, stream>>>(proto, ws_local, ws_counts, out);
}
